// TransformerWithGCN_4758823764052
// MI455X (gfx1250) — compile-verified, matrix operand order checked by reading, not run on hardware
//
#include <hip/hip_runtime.h>
#include <hip/hip_bf16.h>

typedef __attribute__((ext_vector_type(16))) _Float16 v16h;
typedef __attribute__((ext_vector_type(8)))  _Float16 v8h;
typedef __attribute__((ext_vector_type(8)))  float    v8f;
typedef __attribute__((ext_vector_type(4)))  unsigned int u32x4;
typedef __attribute__((ext_vector_type(4)))  int      i32x4;
typedef __attribute__((ext_vector_type(8)))  int      i32x8;

constexpr int S_  = 512;
constexpr int H_  = 768;
constexpr int L_  = 12;
constexpr int NH_ = 12;
constexpr int DH_ = 64;
constexpr int F_  = 3072;
constexpr int GH_ = 128;
constexpr int NC_ = 3;

__device__ __forceinline__ v8f wmma16x16x32(v16h a, v16h b, v8f c) {
  // D = A(16x32 f16) * B(32x16 f16) + C(16x16 f32)
  return __builtin_amdgcn_wmma_f32_16x16x32_f16(false, a, false, b, (short)0, c, false, false);
}

// ---------------------------------------------------------------------------
// TDM: load a 2D tile (tile0 x tile1, 2-byte elements) from global into LDS.
// Descriptor built per CDNA5 ISA 8.3/8.4 (D# group0 128b, group1 256b).
// Must be issued wave-uniformly; tracked by TENSORcnt.
// ---------------------------------------------------------------------------
__device__ __forceinline__ void tdm_load_2d_f16(unsigned lds_off, const void* gaddr,
                                                int dim0, int dim1,
                                                int tile0, int tile1, int stride0)
{
  unsigned long long ga = (unsigned long long)gaddr;
  u32x4 g0;
  g0[0] = 1u;                                              // count=1 (user D#)
  g0[1] = lds_off;                                         // lds_addr (bytes)
  g0[2] = (unsigned)(ga & 0xFFFFFFFFu);                    // global_addr[31:0]
  g0[3] = (unsigned)((ga >> 32) & 0x1FFFFFFu) | (2u << 30);// addr[56:32] | type=2
  i32x8 g1;
  g1[0] = (1 << 16);                                       // data_size=1 (2 bytes)
  g1[1] = (dim0 & 0xFFFF) << 16;                           // tensor_dim0[15:0]
  g1[2] = ((dim0 >> 16) & 0xFFFF) | ((dim1 & 0xFFFF) << 16);
  g1[3] = ((dim1 >> 16) & 0xFFFF) | ((tile0 & 0xFFFF) << 16); // tile_dim0
  g1[4] = (tile1 & 0xFFFF);                                // tile_dim1 (tile_dim2=0)
  g1[5] = stride0;                                         // tensor_dim0_stride[31:0]
  g1[6] = 0;
  g1[7] = 0;
  i32x4 z4 = {0, 0, 0, 0};
#if defined(__clang_major__) && (__clang_major__ >= 23)
  i32x8 z8 = {0, 0, 0, 0, 0, 0, 0, 0};
  __builtin_amdgcn_tensor_load_to_lds(g0, g1, z4, z4, z8, 0);
#else
  __builtin_amdgcn_tensor_load_to_lds(g0, g1, z4, z4, 0);
#endif
}

// ---------------------------------------------------------------------------
// WMMA GEMM, all-f16 operands:  C = act(A16[M,K] * B + bias + resid)
// Bt: B transposed, f16, row-major [Nc,K].
// Block = 4 waves, output 16(M) x 64(N).  Per 128-K stage:
//   - wave 0 TDM-loads the 64x128 B panel into LDS      (TENSORcnt)
//   - all threads async-copy the 16x128 A tile into LDS (ASYNCcnt)
//   - 4 WMMAs per wave, operands from LDS (ds_load_b128)
// K % 128 == 0 required. act: 0 none, 1 relu, 2 exact gelu.
// ---------------------------------------------------------------------------
__global__ void gemm_wmma_kernel(const _Float16* __restrict__ A16,
                                 const _Float16* __restrict__ Bt,
                                 const float* __restrict__ bias,
                                 const float* __restrict__ resid,
                                 float* __restrict__ C32,
                                 _Float16* __restrict__ C16,
                                 int M, int Nc, int K, int act)
{
  __shared__ _Float16 tileA[16][128];   //  4 KB
  __shared__ _Float16 tileB[64][128];   // 16 KB

  const int lane = threadIdx.x;             // 0..31
  const int wave = threadIdx.y;             // 0..3
  const int tid  = wave * 32 + lane;        // 0..127
  int pN = blockIdx.x * 64;
  if (pN + 64 > Nc) pN = Nc - 64;           // block-uniform clamp
  const int tileM = blockIdx.y * 16;
  const int mrow  = lane & 15;
  const int hi    = lane >> 4;

  // per-thread async-copy slot for A: row = tid>>3, two 16B segments
  const int crow = tid >> 3;                // 0..15
  const int cseg = tid & 7;                 // 0..7
  int garow = tileM + crow; if (garow >= M) garow = M - 1;
  const _Float16* gsrcA = A16 + (size_t)garow * K + cseg * 8;
  const unsigned ldsA0 =
      (unsigned)(unsigned long long)(void*)&tileA[crow][cseg * 8];
  const unsigned ldsB =
      (unsigned)(unsigned long long)(void*)&tileB[0][0];

  const _Float16* Bpanel = Bt + (size_t)pN * K;
  const int brow = wave * 16 + mrow;        // B panel row for this lane

  v8f acc = {};
  for (int kc = 0; kc < K; kc += 128) {
    // ---- B panel [64 x 128] via Tensor Data Mover (wave 0 only) ----
    if (wave == 0) {
      tdm_load_2d_f16(ldsB, Bpanel + kc, K, Nc, 128, 64, K);
      __builtin_amdgcn_s_wait_tensorcnt(0);
    }
    // ---- A tile [16 x 128] via async global->LDS (all threads) ----
    {
      unsigned long long ga = (unsigned long long)(const void*)(gsrcA + kc);
      asm volatile("global_load_async_to_lds_b128 %0, %1, off"
                   :: "v"(ldsA0), "v"(ga) : "memory");
      asm volatile("global_load_async_to_lds_b128 %0, %1, off"
                   :: "v"(ldsA0 + 128u), "v"(ga + 128ull) : "memory");
      asm volatile("s_wait_asynccnt 0" ::: "memory");
    }
    __builtin_prefetch(Bpanel + (size_t)(tid & 63) * K + kc + 128, 0, 1);
    __syncthreads();

    #pragma unroll
    for (int ks = 0; ks < 4; ++ks) {
      const int kk = ks * 32;
      v8h a0 = *(const v8h*)&tileA[mrow][kk + hi * 8];
      v8h a1 = *(const v8h*)&tileA[mrow][kk + 16 + hi * 8];
      v8h b0 = *(const v8h*)&tileB[brow][kk + hi * 8];
      v8h b1 = *(const v8h*)&tileB[brow][kk + 16 + hi * 8];
      v16h a, b;
      #pragma unroll
      for (int i = 0; i < 8; ++i) { a[i] = a0[i]; a[8 + i] = a1[i];
                                    b[i] = b0[i]; b[8 + i] = b1[i]; }
      acc = wmma16x16x32(a, b, acc);
    }
    __syncthreads();
  }

  const int c = pN + wave * 16 + mrow;
  const float bv = bias ? bias[c] : 0.0f;
  #pragma unroll
  for (int vv = 0; vv < 8; ++vv) {
    const int r = tileM + vv + hi * 8;
    if (r < M) {
      float val = acc[vv] + bv;
      if (resid) val += resid[(size_t)r * Nc + c];
      if (act == 1)      val = fmaxf(val, 0.0f);
      else if (act == 2) val = 0.5f * val * (1.0f + erff(val * 0.70710678f));
      if (C32) C32[(size_t)r * Nc + c] = val;
      if (C16) C16[(size_t)r * Nc + c] = (_Float16)val;
    }
  }
}

// f32 [R,C] -> f16 transposed [C,R]
__global__ void cvt_transpose_kernel(const float* __restrict__ src,
                                     _Float16* __restrict__ dst, int R, int C)
{
  int c = blockIdx.x * blockDim.x + threadIdx.x;
  int r = blockIdx.y * blockDim.y + threadIdx.y;
  if (r < R && c < C) dst[(size_t)c * R + r] = (_Float16)src[(size_t)r * C + c];
}

// f32 -> f16 elementwise
__global__ void cvt_f16_kernel(const float* __restrict__ src,
                               _Float16* __restrict__ dst, size_t n)
{
  size_t i = (size_t)blockIdx.x * blockDim.x + threadIdx.x;
  if (i < n) dst[i] = (_Float16)src[i];
}

// ---------------------------------------------------------------------------
// Fused attention: one wave per (head, 16-query tile); f16 Q/K/V.
// ---------------------------------------------------------------------------
__global__ void attn_kernel(const _Float16* __restrict__ q,
                            const _Float16* __restrict__ k,
                            const _Float16* __restrict__ vmat,
                            const int* __restrict__ mask,
                            _Float16* __restrict__ ctx16)
{
  __shared__ float     sc[16][S_];
  __shared__ _Float16 p16[16][S_];

  const int lane = threadIdx.x;
  const int head = blockIdx.y;
  const int q0   = blockIdx.x * 16;
  const int mrow = lane & 15;
  const int hi   = lane >> 4;

  const _Float16* Qrow = q + (size_t)(q0 + mrow) * H_ + head * DH_;

  // ---- scores = Q K^T / 8 + mask-bias ----
  for (int kt = 0; kt < S_ / 16; ++kt) {
    const _Float16* Krow = k + (size_t)(kt * 16 + mrow) * H_ + head * DH_;
    v8f acc = {};
    #pragma unroll
    for (int kk = 0; kk < DH_; kk += 32) {
      v8h a0 = *(const v8h*)(Qrow + kk + hi * 8);
      v8h a1 = *(const v8h*)(Qrow + kk + 16 + hi * 8);
      v8h b0 = *(const v8h*)(Krow + kk + hi * 8);
      v8h b1 = *(const v8h*)(Krow + kk + 16 + hi * 8);
      v16h a, b;
      #pragma unroll
      for (int i = 0; i < 8; ++i) { a[i] = a0[i]; a[8 + i] = a1[i];
                                    b[i] = b0[i]; b[8 + i] = b1[i]; }
      acc = wmma16x16x32(a, b, acc);
    }
    const int c = kt * 16 + mrow;
    const float mb = (1.0f - (float)mask[c]) * -1e4f;
    #pragma unroll
    for (int vv = 0; vv < 8; ++vv)
      sc[vv + hi * 8][c] = acc[vv] * 0.125f + mb;
  }
  __syncthreads();

  // ---- softmax per query row (lanes 0..15) ----
  if (lane < 16) {
    float mx = -1e30f;
    for (int j = 0; j < S_; ++j) mx = fmaxf(mx, sc[lane][j]);
    float sum = 0.0f;
    for (int j = 0; j < S_; ++j) { float e = __expf(sc[lane][j] - mx); sc[lane][j] = e; sum += e; }
    const float inv = 1.0f / sum;
    for (int j = 0; j < S_; ++j) p16[lane][j] = (_Float16)(sc[lane][j] * inv);
  }
  __syncthreads();

  // ---- ctx = P @ V ----
  for (int nt = 0; nt < DH_ / 16; ++nt) {
    v8f acc = {};
    for (int kk = 0; kk < S_; kk += 32) {
      v8h a0 = *(const v8h*)&p16[mrow][kk + hi * 8];
      v8h a1 = *(const v8h*)&p16[mrow][kk + 16 + hi * 8];
      v16h a, b;
      #pragma unroll
      for (int i = 0; i < 8; ++i) {
        a[i]     = a0[i];
        a[8 + i] = a1[i];
        b[i]     = vmat[(size_t)(kk + hi * 8 + i)      * H_ + head * DH_ + nt * 16 + mrow];
        b[8 + i] = vmat[(size_t)(kk + 16 + hi * 8 + i) * H_ + head * DH_ + nt * 16 + mrow];
      }
      acc = wmma16x16x32(a, b, acc);
    }
    #pragma unroll
    for (int vv = 0; vv < 8; ++vv)
      ctx16[(size_t)(q0 + vv + hi * 8) * H_ + head * DH_ + nt * 16 + mrow] = (_Float16)acc[vv];
  }
}

// ---------------------------------------------------------------------------
// LayerNorm; optional f16 shadow copy of the output
// ---------------------------------------------------------------------------
__global__ void ln_kernel(const float* __restrict__ x, const float* __restrict__ s,
                          const float* __restrict__ b, float* __restrict__ out32,
                          _Float16* __restrict__ out16, int cols)
{
  __shared__ float red[256];
  const int row = blockIdx.x;
  const int tid = threadIdx.x;
  const float* xr = x + (size_t)row * cols;

  float sum = 0.0f;
  for (int j = tid; j < cols; j += 256) sum += xr[j];
  red[tid] = sum; __syncthreads();
  for (int st = 128; st > 0; st >>= 1) { if (tid < st) red[tid] += red[tid + st]; __syncthreads(); }
  const float mean = red[0] / cols;
  __syncthreads();

  float vs = 0.0f;
  for (int j = tid; j < cols; j += 256) { float d = xr[j] - mean; vs += d * d; }
  red[tid] = vs; __syncthreads();
  for (int st = 128; st > 0; st >>= 1) { if (tid < st) red[tid] += red[tid + st]; __syncthreads(); }
  const float rstd = rsqrtf(red[0] / cols + 1e-12f);

  for (int j = tid; j < cols; j += 256) {
    float val = (xr[j] - mean) * rstd * s[j] + b[j];
    out32[(size_t)row * cols + j] = val;
    if (out16) out16[(size_t)row * cols + j] = (_Float16)val;
  }
}

__global__ void embed_kernel(const int* __restrict__ ids, const int* __restrict__ tt,
                             const float* __restrict__ we, const float* __restrict__ pe,
                             const float* __restrict__ te, float* __restrict__ out)
{
  const int srow = blockIdx.x;
  for (int c = threadIdx.x; c < H_; c += 256)
    out[(size_t)srow * H_ + c] = we[(size_t)ids[srow] * H_ + c]
                               + pe[(size_t)srow * H_ + c]
                               + te[(size_t)tt[srow] * H_ + c];
}

// ---------------- GCN pieces ----------------
__global__ void deg_init(float* deg, int n) {
  int i = blockIdx.x * blockDim.x + threadIdx.x;
  if (i < n) deg[i] = 1.0f;                       // self-loop
}
__global__ void deg_accum(const int* __restrict__ ei, int E, float* deg) {
  int e = blockIdx.x * blockDim.x + threadIdx.x;
  if (e < E) atomicAdd(&deg[ei[(size_t)E + e]], 1.0f);
}
__global__ void dinv_kernel(float* deg, int n) {
  int i = blockIdx.x * blockDim.x + threadIdx.x;
  if (i < n) deg[i] = rsqrtf(deg[i]);             // deg >= 1 always
}
__global__ void scatter_kernel(const int* __restrict__ ei, int E,
                               const float* __restrict__ hw,
                               const float* __restrict__ dinv,
                               float* __restrict__ out)
{
  const int t   = blockIdx.x * blockDim.x + threadIdx.x;
  const int epe = GH_ / 4;                        // 32 threads per edge
  const int e   = t / epe;
  const int c4  = (t % epe) * 4;
  if (e >= E) return;
  const int s = ei[e], d = ei[(size_t)E + e];
  const float nrm = dinv[s] * dinv[d];
  const float4 hv = *(const float4*)(hw + (size_t)s * GH_ + c4);
  float* op = out + (size_t)d * GH_ + c4;
  atomicAdd(op + 0, hv.x * nrm);
  atomicAdd(op + 1, hv.y * nrm);
  atomicAdd(op + 2, hv.z * nrm);
  atomicAdd(op + 3, hv.w * nrm);
}
__global__ void gcn_finish(float* __restrict__ agg, const float* __restrict__ hw,
                           const float* __restrict__ dinv, const float* __restrict__ bias,
                           _Float16* __restrict__ out16, int n, int relu)
{
  const size_t t = (size_t)blockIdx.x * blockDim.x + threadIdx.x;
  const int i = (int)(t / GH_), c = (int)(t % GH_);
  if (i >= n) return;
  float val = agg[t] + hw[t] * dinv[i] * dinv[i] + bias[c];
  if (relu) val = fmaxf(val, 0.0f);
  agg[t] = val;
  if (out16) out16[t] = (_Float16)val;
}
__global__ void pool_kernel(const float* __restrict__ h, int n, float* __restrict__ acc)
{
  const size_t t = (size_t)blockIdx.x * blockDim.x + threadIdx.x;
  if (t < (size_t)n * GH_) atomicAdd(&acc[t % GH_], h[t]);
}

// ---------------- Final combine + classifier ----------------
__global__ void head_kernel(const float* __restrict__ hbert, const float* __restrict__ gp,
                            const float* __restrict__ gh, float invN,
                            const float* __restrict__ combW, const float* __restrict__ combb,
                            const float* __restrict__ clsW,  const float* __restrict__ clsb,
                            float* __restrict__ out)
{
  __shared__ float fc[2 * (H_ + GH_)];
  __shared__ float feat[H_];
  const int tid = threadIdx.x;
  for (int i = tid; i < H_; i += 256) { fc[i] = hbert[i]; fc[H_ + GH_ + i] = hbert[i]; }
  for (int i = tid; i < GH_; i += 256) { fc[H_ + i] = gp[i] * invN; fc[2 * H_ + GH_ + i] = gh[i] * invN; }
  __syncthreads();
  for (int j = tid; j < H_; j += 256) {
    float s = combb[j];
    for (int i = 0; i < 2 * (H_ + GH_); ++i) s += fc[i] * combW[(size_t)i * H_ + j];
    feat[j] = fmaxf(s, 0.0f);
  }
  __syncthreads();
  if (tid < NC_) {
    float s = clsb[tid];
    for (int j = 0; j < H_; ++j) s += feat[j] * clsW[j * NC_ + tid];
    out[tid] = s;
  }
}

// ---------------------------------------------------------------------------
// Host-side launch helpers
// ---------------------------------------------------------------------------
static void run_cvtT(const float* src, _Float16* dst, int R, int C, hipStream_t s) {
  dim3 blk(16, 16), grd((C + 15) / 16, (R + 15) / 16);
  cvt_transpose_kernel<<<grd, blk, 0, s>>>(src, dst, R, C);
}
static void run_gemm(const _Float16* A16, const _Float16* Bt, const float* bias,
                     const float* resid, float* C32, _Float16* C16,
                     int M, int Nc, int K, int act, hipStream_t s) {
  dim3 blk(32, 4), grd((Nc + 63) / 64, (M + 15) / 16);
  gemm_wmma_kernel<<<grd, blk, 0, s>>>(A16, Bt, bias, resid, C32, C16, M, Nc, K, act);
}

static void run_gcn(const float* x, const int* ei, int E, int Nn,
                    const float* W1, const float* b1, const float* W2, const float* b2,
                    _Float16* wT, _Float16* x16, _Float16* g16,
                    float* deg, float* bufA, float* bufB, float* bufC,
                    float* pooled, hipStream_t s)
{
  const int tpb = 256;
  {
    size_t n = (size_t)Nn * H_;
    cvt_f16_kernel<<<(unsigned)((n + tpb - 1) / tpb), tpb, 0, s>>>(x, x16, n);
  }
  deg_init   <<<(Nn + tpb - 1) / tpb, tpb, 0, s>>>(deg, Nn);
  deg_accum  <<<(E  + tpb - 1) / tpb, tpb, 0, s>>>(ei, E, deg);
  dinv_kernel<<<(Nn + tpb - 1) / tpb, tpb, 0, s>>>(deg, Nn);

  // conv1: hw = x @ W1 ; agg = scatter + self-loop + bias -> relu (f32 + f16)
  run_cvtT(W1, wT, H_, GH_, s);
  run_gemm(x16, wT, nullptr, nullptr, bufA, nullptr, Nn, GH_, H_, 0, s);
  hipMemsetAsync(bufB, 0, (size_t)Nn * GH_ * sizeof(float), s);
  {
    size_t tot = (size_t)E * (GH_ / 4);
    scatter_kernel<<<(unsigned)((tot + tpb - 1) / tpb), tpb, 0, s>>>(ei, E, bufA, deg, bufB);
  }
  {
    size_t tot = (size_t)Nn * GH_;
    gcn_finish<<<(unsigned)((tot + tpb - 1) / tpb), tpb, 0, s>>>(bufB, bufA, deg, b1, g16, Nn, 1);
  }

  // conv2
  run_cvtT(W2, wT, GH_, GH_, s);
  run_gemm(g16, wT, nullptr, nullptr, bufA, nullptr, Nn, GH_, GH_, 0, s);
  hipMemsetAsync(bufC, 0, (size_t)Nn * GH_ * sizeof(float), s);
  {
    size_t tot = (size_t)E * (GH_ / 4);
    scatter_kernel<<<(unsigned)((tot + tpb - 1) / tpb), tpb, 0, s>>>(ei, E, bufA, deg, bufC);
  }
  {
    size_t tot = (size_t)Nn * GH_;
    gcn_finish<<<(unsigned)((tot + tpb - 1) / tpb), tpb, 0, s>>>(bufC, bufA, deg, b2, nullptr, Nn, 0);
  }

  // mean pool (sum here; scaled by 1/Nn in head)
  hipMemsetAsync(pooled, 0, GH_ * sizeof(float), s);
  {
    size_t tot = (size_t)Nn * GH_;
    pool_kernel<<<(unsigned)((tot + tpb - 1) / tpb), tpb, 0, s>>>(bufC, Nn, pooled);
  }
}

extern "C" void kernel_launch(void* const* d_in, const int* in_sizes, int n_in,
                              void* d_out, int out_size, void* d_ws, size_t ws_size,
                              hipStream_t stream)
{
  const int*   input_ids = (const int*)d_in[0];
  const int*   attn_mask = (const int*)d_in[1];
  const int*   tok_type  = (const int*)d_in[2];
  const int*   p_edge    = (const int*)d_in[3];
  const int*   h_edge    = (const int*)d_in[4];
  const float* p_x       = (const float*)d_in[5];
  const float* h_x       = (const float*)d_in[6];
  const float* word_emb  = (const float*)d_in[7];
  const float* pos_emb   = (const float*)d_in[8];
  const float* type_emb  = (const float*)d_in[9];
  const float* emb_ln_s  = (const float*)d_in[10];
  const float* emb_ln_b  = (const float*)d_in[11];
  const float* Wq = (const float*)d_in[12]; const float* bq = (const float*)d_in[13];
  const float* Wk = (const float*)d_in[14]; const float* bk = (const float*)d_in[15];
  const float* Wv = (const float*)d_in[16]; const float* bv = (const float*)d_in[17];
  const float* Wo = (const float*)d_in[18]; const float* bo = (const float*)d_in[19];
  const float* ln1_s = (const float*)d_in[20]; const float* ln1_b = (const float*)d_in[21];
  const float* W1 = (const float*)d_in[22]; const float* b1 = (const float*)d_in[23];
  const float* W2 = (const float*)d_in[24]; const float* b2 = (const float*)d_in[25];
  const float* ln2_s = (const float*)d_in[26]; const float* ln2_b = (const float*)d_in[27];
  const float* gcn_W1 = (const float*)d_in[28]; const float* gcn_b1 = (const float*)d_in[29];
  const float* gcn_W2 = (const float*)d_in[30]; const float* gcn_b2 = (const float*)d_in[31];
  const float* comb_W = (const float*)d_in[32]; const float* comb_b = (const float*)d_in[33];
  const float* cls_W  = (const float*)d_in[34]; const float* cls_b  = (const float*)d_in[35];

  const int Ep = in_sizes[3] / 2;
  const int Eh = in_sizes[4] / 2;
  const int Np = in_sizes[5] / H_;
  const int Nh = in_sizes[6] / H_;
  const int Nmax = (Np > Nh) ? Np : Nh;

  // ---- workspace carve ----
  char* ws = (char*)d_ws;
  size_t off = 0;
  auto walloc = [&](size_t bytes) -> void* {
    void* p = ws + off;
    off += (bytes + 255) & ~(size_t)255;
    return p;
  };
  float*    hbuf  = (float*)walloc((size_t)S_ * H_ * 4);
  float*    tmp   = (float*)walloc((size_t)S_ * H_ * 4);
  float*    qb    = (float*)walloc((size_t)S_ * H_ * 4);    // h1 (f32)
  _Float16* h16   = (_Float16*)walloc((size_t)S_ * H_ * 2);
  _Float16* q16   = (_Float16*)walloc((size_t)S_ * H_ * 2); // Q, later h1 (f16)
  _Float16* k16   = (_Float16*)walloc((size_t)S_ * H_ * 2);
  _Float16* v16   = (_Float16*)walloc((size_t)S_ * H_ * 2);
  _Float16* ctx16 = (_Float16*)walloc((size_t)S_ * H_ * 2);
  _Float16* ffn16 = (_Float16*)walloc((size_t)S_ * F_ * 2);
  _Float16* wT    = (_Float16*)walloc((size_t)H_ * F_ * 2);
  _Float16* x16   = (_Float16*)walloc((size_t)Nmax * H_ * 2);
  _Float16* g16   = (_Float16*)walloc((size_t)Nmax * GH_ * 2);
  float*    deg   = (float*)walloc((size_t)Nmax * 4);
  float*    bufA  = (float*)walloc((size_t)Nmax * GH_ * 4);
  float*    bufB  = (float*)walloc((size_t)Nmax * GH_ * 4);
  float*    bufC  = (float*)walloc((size_t)Nmax * GH_ * 4);
  float*    gp    = (float*)walloc(GH_ * 4);
  float*    gh    = (float*)walloc(GH_ * 4);

  // ---- BERT ----
  embed_kernel<<<S_, 256, 0, stream>>>(input_ids, tok_type, word_emb, pos_emb, type_emb, tmp);
  ln_kernel<<<S_, 256, 0, stream>>>(tmp, emb_ln_s, emb_ln_b, hbuf, h16, H_);

  for (int l = 0; l < L_; ++l) {
    const size_t wHH = (size_t)l * H_ * H_;
    const size_t wHF = (size_t)l * H_ * F_;
    // QKV projections (f16 outputs only)
    run_cvtT(Wq + wHH, wT, H_, H_, stream);
    run_gemm(h16, wT, bq + l * H_, nullptr, nullptr, q16, S_, H_, H_, 0, stream);
    run_cvtT(Wk + wHH, wT, H_, H_, stream);
    run_gemm(h16, wT, bk + l * H_, nullptr, nullptr, k16, S_, H_, H_, 0, stream);
    run_cvtT(Wv + wHH, wT, H_, H_, stream);
    run_gemm(h16, wT, bv + l * H_, nullptr, nullptr, v16, S_, H_, H_, 0, stream);
    // fused attention -> ctx16
    attn_kernel<<<dim3(S_ / 16, NH_), 32, 0, stream>>>(q16, k16, v16, attn_mask, ctx16);
    // output projection + residual, LN1 -> (qb f32, q16 f16 reused as h1)
    run_cvtT(Wo + wHH, wT, H_, H_, stream);
    run_gemm(ctx16, wT, bo + l * H_, hbuf, tmp, nullptr, S_, H_, H_, 0, stream);
    ln_kernel<<<S_, 256, 0, stream>>>(tmp, ln1_s + l * H_, ln1_b + l * H_, qb, q16, H_);
    // FFN: gelu(h1 @ W1 + b1) @ W2 + b2 + h1, LN2 -> (hbuf, h16)
    run_cvtT(W1 + wHF, wT, H_, F_, stream);
    run_gemm(q16, wT, b1 + l * F_, nullptr, nullptr, ffn16, S_, F_, H_, 2, stream);
    run_cvtT(W2 + wHF, wT, F_, H_, stream);
    run_gemm(ffn16, wT, b2 + l * H_, qb, tmp, nullptr, S_, H_, F_, 0, stream);
    ln_kernel<<<S_, 256, 0, stream>>>(tmp, ln2_s + l * H_, ln2_b + l * H_, hbuf, h16, H_);
  }

  // ---- GCN branches ----
  run_gcn(p_x, p_edge, Ep, Np, gcn_W1, gcn_b1, gcn_W2, gcn_b2,
          wT, x16, g16, deg, bufA, bufB, bufC, gp, stream);
  run_gcn(h_x, h_edge, Eh, Nh, gcn_W1, gcn_b1, gcn_W2, gcn_b2,
          wT, x16, g16, deg, bufA, bufB, bufC, gh, stream);

  // ---- combine + classifier (CLS = hbuf row 0) ----
  head_kernel<<<1, 256, 0, stream>>>(hbuf, gp, gh, 1.0f / (float)Np,
                                     comb_W, comb_b, cls_W, cls_b, (float*)d_out);
}
